// Low_Rank_GRU_76020921140339
// MI455X (gfx1250) — matrix operand order repaired and checked by reading, not compiled, not run on hardware
//
#include <hip/hip_runtime.h>

// ---------------------------------------------------------------------------
// Low-rank GRU for MI455X (gfx1250): bf16 WMMA everywhere, state in registers.
// Round 2: software-pipelined fragment loads (break s_wait_dscnt 0x0 chains),
// hoisted biases, early-issued global loads.
// ---------------------------------------------------------------------------

typedef __attribute__((ext_vector_type(16))) __bf16          bf16x16;
typedef __attribute__((ext_vector_type(16))) unsigned short  u16x16;
typedef __attribute__((ext_vector_type(8)))  float           f32x8;

constexpr int Bc = 256, Sc = 512, Hc = 512, Ic = 128, Oc = 64, Rc = 8;
constexpr float TAUc = 0.2f, NSc = 0.05f;

// ---- workspace layout (units: unsigned short / bf16 elements) --------------
constexpr size_t N_UB = (size_t)Bc * Sc * Ic;          // u in bf16
constexpr size_t O_UB = 0;
constexpr size_t O_WT = O_UB + N_UB;                   // Wt[3][I][H]
constexpr size_t N_WT = (size_t)3 * Ic * Hc;
constexpr size_t O_NC = O_WT + N_WT;                   // Ncat[H][16] = [N_hr|N_hz]
constexpr size_t N_NC = (size_t)Hc * 16;
constexpr size_t O_NH = O_NC + N_NC;                   // Nh[H][16]   = [N_hh|0]
constexpr size_t O_MR = O_NH + N_NC;                   // Mr[8][H] = M_hr^T
constexpr size_t N_M  = (size_t)8 * Hc;
constexpr size_t O_MZ = O_MR + N_M;                    // Mz[8][H]
constexpr size_t O_MH = O_MZ + N_M;                    // Mh[8][H]
constexpr size_t O_WO = O_MH + N_M;                    // WoutT[H][O]
constexpr size_t N_WO = (size_t)Hc * Oc;

// ---- helpers ---------------------------------------------------------------
__device__ __forceinline__ unsigned short f2bf(float f) {
  unsigned int u = __builtin_bit_cast(unsigned int, f);
  u += 0x7FFFu + ((u >> 16) & 1u);                     // round-to-nearest-even
  return (unsigned short)(u >> 16);
}

union FragU { u16x16 v; bf16x16 b; uint4 q[2]; unsigned short s[16]; };

__device__ __forceinline__ f32x8 wmma_bf16(bf16x16 a, bf16x16 b, f32x8 c) {
  return __builtin_amdgcn_wmma_f32_16x16x32_bf16(false, a, false, b,
                                                 (short)0, c, false, false);
}

__device__ __forceinline__ f32x8 zero8() {
  f32x8 z = {0.f, 0.f, 0.f, 0.f, 0.f, 0.f, 0.f, 0.f};
  return z;
}

// A fragment (16x32, 16-bit): lane L<16 -> row M=L, elems0-7:K=k0+half*8+{0..7},
// elems8-15: K=k0+16+half*8+{0..7}   (half = lane>=16)
__device__ __forceinline__ bf16x16 ldA(const unsigned short* __restrict__ base,
                                       int stride, int m, int k0, int half) {
  const unsigned short* p = base + (size_t)m * stride + k0 + half * 8;
  FragU f;
  f.q[0] = *(const uint4*)(p);
  f.q[1] = *(const uint4*)(p + 16);
  return f.b;
}

// B fragment (32x16, 16-bit): lane L -> col N=n0+L%16, elems0-7: K=k0+half*8+{0..7},
// elems8-15: K=k0+16+half*8+{0..7}  (K arg = k0 + L)
__device__ __forceinline__ bf16x16 ldB(const unsigned short* __restrict__ base,
                                       int stride, int K, int n0) {
  const int k0 = K & ~31, l = K & 31;
  const unsigned short* p = base + (size_t)(k0 + (l >> 4) * 8) * stride + n0 + (l & 15);
  FragU f;
#pragma unroll
  for (int j = 0; j < 8; ++j) {
    f.s[j] = p[(size_t)j * stride];
    f.s[8 + j] = p[(size_t)(16 + j) * stride];
  }
  return f.b;
}

// B fragment with only K in [lo,hi) valid; source is compact [hi-lo][stride].
__device__ __forceinline__ bf16x16 ldB_band(const unsigned short* __restrict__ base,
                                            int stride, int K, int n0,
                                            int lo, int hi) {
  const int k0 = K & ~31, l = K & 31;
  const int n = n0 + (l & 15), kb = k0 + (l >> 4) * 8;
  FragU f;
#pragma unroll
  for (int s = 0; s < 16; ++s) {
    int k = kb + (s < 8 ? s : 16 + s - 8);
    bool ok = (k >= lo) && (k < hi);
    f.s[s] = ok ? base[(size_t)(k - lo) * stride + n] : (unsigned short)0;
  }
  return f.b;
}

// Build A fragment of v (16x32, valid K = 0..15) from f32 LDS buffer vb[16][16].
__device__ __forceinline__ bf16x16 buildVA(const float (*vb)[16], int m, int half) {
  FragU f;
  f.q[0] = make_uint4(0, 0, 0, 0);
  f.q[1] = make_uint4(0, 0, 0, 0);
#pragma unroll
  for (int j = 0; j < 8; ++j) f.s[j] = f2bf(vb[m][half * 8 + j]);
  return f.b;
}

// ---------------------------------------------------------------------------
// Kernel 0: convert/pack weights and u into bf16 workspace layouts.
// ---------------------------------------------------------------------------
__global__ void gru_prep_kernel(const float* __restrict__ u,
                                const float* __restrict__ Mhr, const float* __restrict__ Nhr,
                                const float* __restrict__ Mhz, const float* __restrict__ Nhz,
                                const float* __restrict__ Mhh, const float* __restrict__ Nhh,
                                const float* __restrict__ Wir, const float* __restrict__ Wiz,
                                const float* __restrict__ Wih, const float* __restrict__ Wout,
                                unsigned short* __restrict__ ws) {
  const size_t g0 = (size_t)blockIdx.x * blockDim.x + threadIdx.x;
  const size_t gs = (size_t)gridDim.x * blockDim.x;
  unsigned short* ub = ws + O_UB;
  unsigned short* wt = ws + O_WT;
  unsigned short* nc = ws + O_NC;
  unsigned short* nh = ws + O_NH;
  unsigned short* mr = ws + O_MR;
  unsigned short* mz = ws + O_MZ;
  unsigned short* mh = ws + O_MH;
  unsigned short* wo = ws + O_WO;

  for (size_t i = g0; i < N_UB; i += gs) ub[i] = f2bf(u[i]);
  for (size_t i = g0; i < N_WT; i += gs) {
    int g = (int)(i / ((size_t)Ic * Hc));
    int rem = (int)(i % ((size_t)Ic * Hc));
    int k = rem / Hc, h = rem % Hc;
    const float* W = (g == 0) ? Wir : ((g == 1) ? Wiz : Wih);
    wt[i] = f2bf(W[(size_t)h * Ic + k]);                 // Wt[g][i][h] = W[h][i]
  }
  for (size_t i = g0; i < N_NC; i += gs) {
    int h = (int)(i / 16), j = (int)(i % 16);
    nc[i] = f2bf((j < Rc) ? Nhr[(size_t)h * Rc + j] : Nhz[(size_t)h * Rc + (j - Rc)]);
    nh[i] = (j < Rc) ? f2bf(Nhh[(size_t)h * Rc + j]) : (unsigned short)0;
  }
  for (size_t i = g0; i < N_M; i += gs) {
    int j = (int)(i / Hc), h = (int)(i % Hc);
    mr[i] = f2bf(Mhr[(size_t)h * Rc + j]);               // M^T[j][h]
    mz[i] = f2bf(Mhz[(size_t)h * Rc + j]);
    mh[i] = f2bf(Mhh[(size_t)h * Rc + j]);
  }
  for (size_t i = g0; i < N_WO; i += gs) {
    int h = (int)(i / Oc), o = (int)(i % Oc);
    wo[i] = f2bf(Wout[(size_t)o * Hc + h]);              // Wout^T[h][o]
  }
}

// ---------------------------------------------------------------------------
// Kernel 1: fused recurrent scan. 16 blocks x 256 threads; block owns 16 batch
// rows; wave w owns h-tiles [w*64, w*64+64). State lives in C-layout VGPRs.
// All fragment loads are software-pipelined one stage deep.
// ---------------------------------------------------------------------------
__global__ __launch_bounds__(256) void gru_scan_kernel(
    const unsigned short* __restrict__ ub,   // [B][S][I] bf16
    const unsigned short* __restrict__ Wt,   // [3][I][H] bf16
    const unsigned short* __restrict__ Nc,   // [H][16]
    const unsigned short* __restrict__ Nh,   // [H][16]
    const unsigned short* __restrict__ Mr,   // [8][H]
    const unsigned short* __restrict__ Mz,   // [8][H]
    const unsigned short* __restrict__ Mh,   // [8][H]
    const float* __restrict__ bir, const float* __restrict__ biz,
    const float* __restrict__ bih,
    const float* __restrict__ x0, const float* __restrict__ noise,
    float* __restrict__ traj, float* __restrict__ xfin) {
  __shared__ unsigned short xbuf[16][Hc];    // bf16 mirror of state / r*x
  __shared__ unsigned short NcL[Hc][16];
  __shared__ unsigned short NhL[Hc][16];
  __shared__ float vbuf[16][16];             // v exchange (C -> A transpose)

  const int tid = threadIdx.x;
  const int wave = tid >> 5, lane = tid & 31;
  const int b0 = blockIdx.x * 16;
  const int mrow = lane & 15;                // A-row index for this lane
  const int half = lane >> 4;
  const int ncol = lane & 15;                // C-layout column index

  const unsigned short* Wt0 = Wt;
  const unsigned short* Wt1 = Wt + (size_t)Ic * Hc;
  const unsigned short* Wt2 = Wt + 2 * (size_t)Ic * Hc;

  {  // stage N panels into LDS (reused 8x per step by all waves)
    const uint* s1 = (const uint*)Nc;
    const uint* s2 = (const uint*)Nh;
    uint* d1 = (uint*)&NcL[0][0];
    uint* d2 = (uint*)&NhL[0][0];
    for (int i = tid; i < Hc * 16 / 2; i += 256) { d1[i] = s1[i]; d2[i] = s2[i]; }
  }

  // time-invariant per-lane biases (hoisted out of the scan)
  float brv[4], bzv[4], bhv[4];
#pragma unroll
  for (int tt = 0; tt < 4; ++tt) {
    int h = (wave * 4 + tt) * 16 + ncol;
    brv[tt] = bir[h];
    bzv[tt] = biz[h];
    bhv[tt] = bih[h];
  }

  f32x8 x[4];
#pragma unroll
  for (int tt = 0; tt < 4; ++tt) {           // load x0 into C-layout frags
    int h = (wave * 4 + tt) * 16 + ncol;
    f32x8 v;
#pragma unroll
    for (int r = 0; r < 8; ++r) {
      int m = r + 8 * half;
      v[r] = x0[(size_t)(b0 + m) * Hc + h];
      xbuf[m][h] = f2bf(v[r]);
    }
    x[tt] = v;
  }
  __syncthreads();

  for (int t = 0; t < Sc; ++t) {
    // ---- early issue of long-latency global loads for this step -----------
    const unsigned short* ubase = ub + ((size_t)(b0 + mrow) * Sc + t) * Ic;
    if (t + 1 < Sc)
      __builtin_prefetch(ubase + Ic, 0, 1);  // -> global_prefetch_b8
    bf16x16 uA[4];
#pragma unroll
    for (int kk = 0; kk < 4; ++kk) uA[kk] = ldA(ubase, 0, 0, kk * 32, half);

    f32x8 nzv[4];
#pragma unroll
    for (int tt = 0; tt < 4; ++tt) {
      int h = (wave * 4 + tt) * 16 + ncol;
#pragma unroll
      for (int r = 0; r < 8; ++r) {
        int m = r + 8 * half;
        nzv[tt][r] = noise[((size_t)t * Bc + (b0 + m)) * Hc + h];
      }
    }

    // ---- (a) v_rz = (x @ [N_hr|N_hz]) : K=512, pipelined LDS GEMM ---------
    f32x8 vrz = zero8();
    {
      bf16x16 Acur = ldA(&xbuf[0][0], Hc, mrow, 0, half);
      bf16x16 Bcur = ldB(&NcL[0][0], 16, lane, 0);
#pragma unroll
      for (int kk = 0; kk < 16; ++kk) {
        bf16x16 Anxt = Acur, Bnxt = Bcur;
        if (kk < 15) {
          Anxt = ldA(&xbuf[0][0], Hc, mrow, (kk + 1) * 32, half);
          Bnxt = ldB(&NcL[0][0], 16, (kk + 1) * 32 + lane, 0);
        }
        vrz = wmma_bf16(Acur, Bcur, vrz);
        Acur = Anxt; Bcur = Bnxt;
      }
    }
    if (wave == 0) {
#pragma unroll
      for (int r = 0; r < 8; ++r)
        vbuf[r + 8 * half][ncol] = vrz[r] * (1.0f / (float)Hc);
    }
    __syncthreads();

    bf16x16 vA = buildVA(vbuf, mrow, half);

    // ---- input projections + r/z gates, flat pipelined over (tile,kk) -----
    f32x8 zf[4], ihp[4];
    {
      const int h0f = wave * 64;
      bf16x16 Br = ldB(Wt0, Hc, lane, h0f);
      bf16x16 Bz = ldB(Wt1, Hc, lane, h0f);
      bf16x16 Bh = ldB(Wt2, Hc, lane, h0f);
      bf16x16 MrF = ldB_band(Mr, Hc, lane, h0f, 0, 8);   // for tile 0 epilogue
      bf16x16 MzF = ldB_band(Mz, Hc, lane, h0f, 8, 16);
      f32x8 pr = zero8(), pz = zero8(), ph = zero8();
#pragma unroll
      for (int it = 0; it < 16; ++it) {
        const int tt = it >> 2, kk = it & 3;
        // preload next trio (and next tile's M fragments at tile boundary)
        bf16x16 BrN = Br, BzN = Bz, BhN = Bh, MrN = MrF, MzN = MzF;
        if (it < 15) {
          const int itn = it + 1;
          const int ttn = itn >> 2, kkn = itn & 3;
          const int h0n = (wave * 4 + ttn) * 16;
          const int Kn = kkn * 32 + lane;
          BrN = ldB(Wt0, Hc, Kn, h0n);
          BzN = ldB(Wt1, Hc, Kn, h0n);
          BhN = ldB(Wt2, Hc, Kn, h0n);
          if (kkn == 0) {                     // entering a new tile next iter
            MrN = ldB_band(Mr, Hc, lane, h0n, 0, 8);
            MzN = ldB_band(Mz, Hc, lane, h0n, 8, 16);
          }
        }
        pr = wmma_bf16(uA[kk], Br, pr);
        pz = wmma_bf16(uA[kk], Bz, pz);
        ph = wmma_bf16(uA[kk], Bh, ph);
        if (kk == 3) {                        // tile epilogue
          const int h0 = (wave * 4 + tt) * 16;
          f32x8 hr = wmma_bf16(vA, MrF, zero8());
          f32x8 hz = wmma_bf16(vA, MzF, zero8());
#pragma unroll
          for (int r = 0; r < 8; ++r) {
            float rr = 1.f / (1.f + __expf(-(pr[r] + brv[tt] + hr[r])));
            float zz = 1.f / (1.f + __expf(-(pz[r] + bzv[tt] + hz[r])));
            zf[tt][r] = zz;
            xbuf[r + 8 * half][h0 + ncol] = f2bf(rr * x[tt][r]);  // r*x -> LDS
          }
          ihp[tt] = ph;
          pr = zero8(); pz = zero8(); ph = zero8();
        }
        Br = BrN; Bz = BzN; Bh = BhN; MrF = MrN; MzF = MzN;
      }
    }
    __syncthreads();

    // ---- preload M_hh^T fragments for all 4 tiles (hidden under v_h GEMM) -
    bf16x16 MhF[4];
#pragma unroll
    for (int tt = 0; tt < 4; ++tt)
      MhF[tt] = ldB_band(Mh, Hc, lane, (wave * 4 + tt) * 16, 0, 8);

    // ---- (e) v_h = ((r*x) @ N_hh) : K=512, pipelined LDS GEMM -------------
    f32x8 vh = zero8();
    {
      bf16x16 Acur = ldA(&xbuf[0][0], Hc, mrow, 0, half);
      bf16x16 Bcur = ldB(&NhL[0][0], 16, lane, 0);
#pragma unroll
      for (int kk = 0; kk < 16; ++kk) {
        bf16x16 Anxt = Acur, Bnxt = Bcur;
        if (kk < 15) {
          Anxt = ldA(&xbuf[0][0], Hc, mrow, (kk + 1) * 32, half);
          Bnxt = ldB(&NhL[0][0], 16, (kk + 1) * 32 + lane, 0);
        }
        vh = wmma_bf16(Acur, Bcur, vh);
        Acur = Anxt; Bcur = Bnxt;
      }
    }
    if (wave == 0) {
#pragma unroll
      for (int r = 0; r < 8; ++r)
        vbuf[r + 8 * half][ncol] = vh[r] * (1.0f / (float)Hc);
    }
    __syncthreads();

    bf16x16 vAh = buildVA(vbuf, mrow, half);

#pragma unroll
    for (int tt = 0; tt < 4; ++tt) {
      const int h0 = (wave * 4 + tt) * 16;
      const int h = h0 + ncol;
      f32x8 hh = wmma_bf16(vAh, MhF[tt], zero8());
      f32x8 xn;
#pragma unroll
      for (int r = 0; r < 8; ++r) {
        int m = r + 8 * half;
        float g = tanhf(ihp[tt][r] + bhv[tt] + hh[r]);
        float xc = x[tt][r];
        float rec = (1.f - zf[tt][r]) * g + zf[tt][r] * xc;
        float v = xc + NSc * nzv[tt][r] + TAUc * (rec - xc);
        xn[r] = v;
        traj[((size_t)(b0 + m) * Sc + t) * Hc + h] = v;
      }
      x[tt] = xn;
#pragma unroll
      for (int r = 0; r < 8; ++r) xbuf[r + 8 * half][h] = f2bf(xn[r]);
    }
    __syncthreads();
  }

#pragma unroll
  for (int tt = 0; tt < 4; ++tt) {
    int h = (wave * 4 + tt) * 16 + ncol;
#pragma unroll
    for (int r = 0; r < 8; ++r) {
      int m = r + 8 * half;
      xfin[(size_t)(b0 + m) * Hc + h] = x[tt][r];
    }
  }
}

// ---------------------------------------------------------------------------
// Kernel 2: output projection  out[b,s,:] = traj[b,s,:] @ Wout^T + b
// 1024 blocks x 8 waves; each wave: one 16-row M-tile, all 4 N-tiles, K=512.
// A (f32 traj) and B loads pipelined one K-chunk ahead.
// ---------------------------------------------------------------------------
__global__ __launch_bounds__(256) void gru_outproj_kernel(
    const float* __restrict__ traj, const unsigned short* __restrict__ WoT,
    const float* __restrict__ bo, float* __restrict__ out) {
  const int tid = threadIdx.x;
  const int wave = tid >> 5, lane = tid & 31;
  const int mt = blockIdx.x * 8 + wave;      // 0 .. B*S/16-1
  const int row0 = mt * 16;
  const int mrow = lane & 15, half = lane >> 4, ncol = lane & 15;

  f32x8 acc[4];
#pragma unroll
  for (int nt = 0; nt < 4; ++nt) acc[nt] = zero8();

  const float* abase = traj + (size_t)(row0 + mrow) * Hc + half * 8;

  float4 a0 = *(const float4*)(abase + 0);
  float4 a1 = *(const float4*)(abase + 4);
  float4 a2 = *(const float4*)(abase + 16);
  float4 a3 = *(const float4*)(abase + 20);
  bf16x16 Bn[4];
#pragma unroll
  for (int nt = 0; nt < 4; ++nt) Bn[nt] = ldB(WoT, Oc, lane, nt * 16);

#pragma unroll
  for (int kk = 0; kk < 16; ++kk) {
    float4 n0 = a0, n1 = a1, n2 = a2, n3 = a3;
    bf16x16 Bc4[4];
#pragma unroll
    for (int nt = 0; nt < 4; ++nt) Bc4[nt] = Bn[nt];
    if (kk < 15) {
      const float* p = abase + (kk + 1) * 32;
      n0 = *(const float4*)(p + 0);
      n1 = *(const float4*)(p + 4);
      n2 = *(const float4*)(p + 16);
      n3 = *(const float4*)(p + 20);
#pragma unroll
      for (int nt = 0; nt < 4; ++nt)
        Bn[nt] = ldB(WoT, Oc, (kk + 1) * 32 + lane, nt * 16);
    }
    FragU f;
    f.s[0] = f2bf(a0.x); f.s[1] = f2bf(a0.y); f.s[2] = f2bf(a0.z); f.s[3] = f2bf(a0.w);
    f.s[4] = f2bf(a1.x); f.s[5] = f2bf(a1.y); f.s[6] = f2bf(a1.z); f.s[7] = f2bf(a1.w);
    f.s[8] = f2bf(a2.x); f.s[9] = f2bf(a2.y); f.s[10] = f2bf(a2.z); f.s[11] = f2bf(a2.w);
    f.s[12] = f2bf(a3.x); f.s[13] = f2bf(a3.y); f.s[14] = f2bf(a3.z); f.s[15] = f2bf(a3.w);
    bf16x16 A = f.b;
#pragma unroll
    for (int nt = 0; nt < 4; ++nt) acc[nt] = wmma_bf16(A, Bc4[nt], acc[nt]);
    a0 = n0; a1 = n1; a2 = n2; a3 = n3;
  }
#pragma unroll
  for (int nt = 0; nt < 4; ++nt) {
    int o = nt * 16 + ncol;
    float bb = bo[o];
#pragma unroll
    for (int r = 0; r < 8; ++r) {
      int m = row0 + r + 8 * half;
      out[(size_t)m * Oc + o] = acc[nt][r] + bb;
    }
  }
}

// ---------------------------------------------------------------------------
extern "C" void kernel_launch(void* const* d_in, const int* in_sizes, int n_in,
                              void* d_out, int out_size, void* d_ws, size_t ws_size,
                              hipStream_t stream) {
  const float* u     = (const float*)d_in[0];
  const float* x0    = (const float*)d_in[1];
  const float* noise = (const float*)d_in[2];
  const float* Mhr   = (const float*)d_in[3];
  const float* Nhr   = (const float*)d_in[4];
  const float* Mhz   = (const float*)d_in[5];
  const float* Nhz   = (const float*)d_in[6];
  const float* Mhh   = (const float*)d_in[7];
  const float* Nhh   = (const float*)d_in[8];
  const float* Wir_w = (const float*)d_in[9];
  const float* Wir_b = (const float*)d_in[10];
  const float* Wiz_w = (const float*)d_in[11];
  const float* Wiz_b = (const float*)d_in[12];
  const float* Wih_w = (const float*)d_in[13];
  const float* Wih_b = (const float*)d_in[14];
  const float* Wout_w = (const float*)d_in[15];
  const float* Wout_b = (const float*)d_in[16];

  float* out  = (float*)d_out;                       // [B][S][O]
  float* xfin = out + (size_t)Bc * Sc * Oc;          // [B][H]
  float* traj = xfin + (size_t)Bc * Hc;              // [B][S][H]
  unsigned short* ws = (unsigned short*)d_ws;

  gru_prep_kernel<<<4096, 256, 0, stream>>>(u, Mhr, Nhr, Mhz, Nhz, Mhh, Nhh,
                                            Wir_w, Wiz_w, Wih_w, Wout_w, ws);

  gru_scan_kernel<<<Bc / 16, 256, 0, stream>>>(
      ws + O_UB, ws + O_WT, ws + O_NC, ws + O_NH, ws + O_MR, ws + O_MZ,
      ws + O_MH, Wir_b, Wiz_b, Wih_b, x0, noise, traj, xfin);

  gru_outproj_kernel<<<(Bc * Sc / 16) / 8, 256, 0, stream>>>(traj, ws + O_WO,
                                                             Wout_b, out);
}
